// EABranch_31447750541442
// MI455X (gfx1250) — matrix operand order repaired and checked by reading, not compiled, not run on hardware
//
#include <hip/hip_runtime.h>
#include <hip/hip_bf16.h>

// Problem constants (match reference)
#define B_    8
#define CH_   128
#define CL_   256
#define HH_   64
#define WH_   64
#define HWH_  (HH_ * WH_)
#define HWL_  (128 * 128)   // 16384
#define K_    64
#define EPS_  1e-5f

typedef __attribute__((ext_vector_type(16))) __bf16 v16bf;
typedef __attribute__((ext_vector_type(8)))  __bf16 v8bf;
typedef __attribute__((ext_vector_type(8)))  float  v8f;

// ---------------------------------------------------------------------------
// Kernel 1: fold BN params into scale/shift for both branches
// ---------------------------------------------------------------------------
__global__ void ea_prep(const float* __restrict__ g_h, const float* __restrict__ b_h,
                        const float* __restrict__ m_h, const float* __restrict__ v_h,
                        const float* __restrict__ g_l, const float* __restrict__ b_l,
                        const float* __restrict__ m_l, const float* __restrict__ v_l,
                        float* __restrict__ scale_h, float* __restrict__ shift_h,
                        float* __restrict__ scale_l, float* __restrict__ shift_l) {
  int i = threadIdx.x;
  if (i < CL_) {
    float s = g_l[i] / sqrtf(v_l[i] + EPS_);
    scale_l[i] = s;
    shift_l[i] = b_l[i] - m_l[i] * s;
  }
  if (i < CH_) {
    float s = g_h[i] / sqrtf(v_h[i] + EPS_);
    scale_h[i] = s;
    shift_h[i] = b_h[i] - m_h[i] * s;
  }
}

// ---------------------------------------------------------------------------
// Kernel 2: BN(x_h) + 8x8 adaptive max-pool -> p[b][c][k], k = 8x8 = 64
// ---------------------------------------------------------------------------
__global__ void ea_pool(const float* __restrict__ x_h,
                        const float* __restrict__ scale_h,
                        const float* __restrict__ shift_h,
                        float* __restrict__ p) {
  int tid = blockIdx.x * blockDim.x + threadIdx.x;   // 65536 total
  int k = tid & 63;
  int c = (tid >> 6) & (CH_ - 1);
  int b = tid >> 13;
  int ki = k >> 3, kj = k & 7;
  const float* src = x_h + (size_t)(b * CH_ + c) * HWH_;
  float sc = scale_h[c], sh = shift_h[c];
  float m = -3.0e38f;
  for (int dh = 0; dh < 8; ++dh) {
    const float* row = src + (ki * 8 + dh) * WH_ + kj * 8;
#pragma unroll
    for (int dw = 0; dw < 8; ++dw)
      m = fmaxf(m, fmaf(row[dw], sc, sh));
  }
  p[tid] = m;   // tid == ((b*CH_+c)*K_)+k
}

// ---------------------------------------------------------------------------
// Kernel 3: kv projection (512x128 x 128x64 per batch), emit bf16 operands.
// BN-scale of x_l is folded into the GEMM1 A operand:
//   ckb[b][k][c] = bf16( kv[b][o=c][k] * scale_l[c] )   (c < 256)
//   ckf[b][k][c] = f32 ( kv[b][o=c][k] )                (for bias kernel)
//   cvb[b][c][k] = bf16( kv[b][o=256+c][k] )            -- GEMM2 A operand
// ---------------------------------------------------------------------------
__global__ void ea_kv(const float* __restrict__ p,
                      const float* __restrict__ kv_w,
                      const float* __restrict__ kv_b,
                      const float* __restrict__ scale_l,
                      __bf16* __restrict__ ckb,
                      float*  __restrict__ ckf,
                      __bf16* __restrict__ cvb) {
  int tid = blockIdx.x * blockDim.x + threadIdx.x;   // 262144 total
  int k = tid & 63;
  int o = (tid >> 6) & 511;
  int b = tid >> 15;
  float acc = kv_b[o];
  const float* pc = p + (size_t)b * CH_ * K_ + k;
  const float* w  = kv_w + (size_t)o * CH_;
  for (int c = 0; c < CH_; ++c)
    acc = fmaf(pc[c * K_], w[c], acc);
  if (o < CL_) {
    ckf[(size_t)(b * K_ + k) * CL_ + o] = acc;
    ckb[(size_t)(b * K_ + k) * CL_ + o] = (__bf16)(acc * scale_l[o]);
  } else {
    cvb[(size_t)(b * CL_ + (o - CL_)) * K_ + k] = (__bf16)acc;
  }
}

// ---------------------------------------------------------------------------
// Kernel 3b: per-(b,k) logit bias = reatt[k] + sum_c ck[k][c]*shift_l[c]
// ---------------------------------------------------------------------------
__global__ void ea_bias(const float* __restrict__ ckf,
                        const float* __restrict__ shift_l,
                        const float* __restrict__ reatt,
                        float* __restrict__ bias) {
  int tid = blockIdx.x * blockDim.x + threadIdx.x;   // B_*K_ = 512 total
  int k = tid & 63;
  const float* row = ckf + (size_t)tid * CL_;
  float acc = reatt[k];
  for (int c = 0; c < CL_; ++c)
    acc = fmaf(row[c], shift_l[c], acc);
  bias[tid] = acc;
}

// ---------------------------------------------------------------------------
// Kernel 4: main attention. Block = 256 threads (8 waves). Each wave handles
// one 16-column spatial tile of one batch.
//   GEMM1: attn(64 x 16) = ck_scaled(64 x 256) * x_l(256 x 16)  [4 Mt x 8 Ks]
//     x_l feed: cooperative — each lane loads one channel row of the 32x16
//     tile as 4x b128, converts with packed cvts into one v16bf, stores it
//     contiguously to LDS (2x ds_store_b128); the WMMA B fragment is then
//     gathered with 16 D16 LDS loads (the transpose happens on the read side,
//     where d16/d16_hi loads land directly in register halves).
//   softmax over k (bias added; lane pairs via shfl_xor 16)
//   GEMM2: out(256 x 16) = cv(256 x 64) * attn(64 x 16) + x_l   [16 Mt x 2 Ks]
// ---------------------------------------------------------------------------
__global__ void ea_main(const float* __restrict__ x_l,
                        const __bf16* __restrict__ ckb,
                        const __bf16* __restrict__ cvb,
                        const float* __restrict__ bias,
                        float* __restrict__ out) {
  __shared__ __align__(32) __bf16 attn_sm[8 * 16 * K_];   // [wave][n][k], 16KB
  __shared__ __align__(32) __bf16 xstage[8 * 32 * 16];    // [wave][c_loc][n], 8KB

  const int wave = threadIdx.x >> 5;
  const int lane = threadIdx.x & 31;
  const int half = lane >> 4;        // 0: lanes 0-15, 1: lanes 16-31
  const int n    = lane & 15;        // column within tile / A-matrix row
  const int b    = blockIdx.y;
  const int hw0  = blockIdx.x * 128 + wave * 16;
  const int hw   = hw0 + n;

  __bf16* xst = xstage + wave * (32 * 16);   // this wave's staging tile

  // ---------------- GEMM1: attn logits -------------------------------------
  v8f acc[4] = {};   // 4 M-tiles over k (64 rows)
#pragma unroll
  for (int c0 = 0; c0 < CL_; c0 += 32) {
    // Cooperative load: lane owns channel row c = c0 + lane, 16 columns.
    const float* xrow = x_l + ((size_t)(b * CL_ + c0 + lane)) * HWL_ + hw0;
    float4 r0 = *(const float4*)(xrow + 0);
    float4 r1 = *(const float4*)(xrow + 4);
    float4 r2 = *(const float4*)(xrow + 8);
    float4 r3 = *(const float4*)(xrow + 12);
    // Pack the whole row with paired cvts, store contiguously (2x b128).
    v16bf rowbf;
    rowbf[0]  = (__bf16)r0.x;  rowbf[1]  = (__bf16)r0.y;
    rowbf[2]  = (__bf16)r0.z;  rowbf[3]  = (__bf16)r0.w;
    rowbf[4]  = (__bf16)r1.x;  rowbf[5]  = (__bf16)r1.y;
    rowbf[6]  = (__bf16)r1.z;  rowbf[7]  = (__bf16)r1.w;
    rowbf[8]  = (__bf16)r2.x;  rowbf[9]  = (__bf16)r2.y;
    rowbf[10] = (__bf16)r2.z;  rowbf[11] = (__bf16)r2.w;
    rowbf[12] = (__bf16)r3.x;  rowbf[13] = (__bf16)r3.y;
    rowbf[14] = (__bf16)r3.z;  rowbf[15] = (__bf16)r3.w;
    *(v16bf*)(xst + lane * 16) = rowbf;
    __builtin_amdgcn_wave_barrier();   // wave-private staging; order ds ops

    // B fragment: 32x16 bf16, lanes 0-15 K=c0+0..7,16..23, lanes 16-31 K=c0+8..15,24..31.
    // Transposing gather: element e -> row c_loc = (e>>3)*16+half*8+(e&7), column n.
    v16bf bfrag;
#pragma unroll
    for (int e = 0; e < 16; ++e)
      bfrag[e] = xst[((e >> 3) * 16 + half * 8 + (e & 7)) * 16 + n];
    __builtin_amdgcn_wave_barrier();   // keep next chunk's stores after reads

#pragma unroll
    for (int t = 0; t < 4; ++t) {
      // A fragment: row m = t*16 + n over k-index; K(c) split per 16-bit A layout
      const __bf16* arow = ckb + (size_t)(b * K_ + t * 16 + n) * CL_;
      const int coff = c0 + half * 8;
      v8bf lo = *(const v8bf*)(arow + coff);        // K = coff..coff+7
      v8bf hi = *(const v8bf*)(arow + coff + 16);   // K = coff+16..coff+23
      v16bf afrag = __builtin_shufflevector(lo, hi, 0,1,2,3,4,5,6,7,8,9,10,11,12,13,14,15);
      acc[t] = __builtin_amdgcn_wmma_f32_16x16x32_bf16(
          false, afrag, false, bfrag, (short)0, acc[t], false, false);
    }
  }

  // ---------------- softmax over k (64) per column -------------------------
  // lane holds 32 of the 64 k-values of its column; partner = lane^16 holds rest
  const float* brow = bias + b * K_;
  float vals[32];
  float mx = -3.0e38f;
#pragma unroll
  for (int t = 0; t < 4; ++t)
#pragma unroll
    for (int r = 0; r < 8; ++r) {
      const int k = t * 16 + half * 8 + r;
      float v = (acc[t][r] + brow[k]) * 0.125f;   // K^-0.5 = 1/8
      vals[t * 8 + r] = v;
      mx = fmaxf(mx, v);
    }
  mx = fmaxf(mx, __shfl_xor(mx, 16, 32));
  float s = 0.f;
#pragma unroll
  for (int i = 0; i < 32; ++i) {
    float e = __expf(vals[i] - mx);
    vals[i] = e;
    s += e;
  }
  s += __shfl_xor(s, 16, 32);
  const float inv = 1.0f / s;

  // stage normalized attn to LDS as bf16, layout [n][k] (k contiguous)
  __bf16* ab = attn_sm + wave * (16 * K_);
#pragma unroll
  for (int t = 0; t < 4; ++t)
#pragma unroll
    for (int r = 0; r < 8; ++r) {
      const int k = t * 16 + half * 8 + r;
      ab[n * K_ + k] = (__bf16)(vals[t * 8 + r] * inv);
    }
  __syncthreads();

  // ---------------- GEMM2: out = cv * attn + x_l ---------------------------
  // B fragments: one aligned 32B LDS load each (K contiguous per lane)
  v16bf bfr0 = *(const v16bf*)(ab + n * K_ + half * 16);        // K = 0..31 chunk
  v16bf bfr1 = *(const v16bf*)(ab + n * K_ + 32 + half * 16);   // K = 32..63 chunk

  const float* xres = x_l + (size_t)b * CL_ * HWL_ + hw;
  float* outc       = out + (size_t)b * CL_ * HWL_ + hw;

  for (int ct = 0; ct < 16; ++ct) {
    const int c0 = ct * 16;
    // C init = residual (raw x_l), matching 16x16 f32 C/D layout
    v8f cacc;
#pragma unroll
    for (int r = 0; r < 8; ++r)
      cacc[r] = xres[(size_t)(c0 + half * 8 + r) * HWL_];
#pragma unroll
    for (int kc = 0; kc < 2; ++kc) {
      const __bf16* arow = cvb + (size_t)(b * CL_ + c0 + n) * K_;
      const int koff = kc * 32 + half * 16;
      v8bf lo = *(const v8bf*)(arow + koff);
      v8bf hi = *(const v8bf*)(arow + koff + 8);
      v16bf afrag = __builtin_shufflevector(lo, hi, 0,1,2,3,4,5,6,7,8,9,10,11,12,13,14,15);
      cacc = __builtin_amdgcn_wmma_f32_16x16x32_bf16(
          false, afrag, false, (kc == 0 ? bfr0 : bfr1), (short)0, cacc, false, false);
    }
#pragma unroll
    for (int r = 0; r < 8; ++r)
      outc[(size_t)(c0 + half * 8 + r) * HWL_] = cacc[r];
  }
}

// ---------------------------------------------------------------------------
extern "C" void kernel_launch(void* const* d_in, const int* in_sizes, int n_in,
                              void* d_out, int out_size, void* d_ws, size_t ws_size,
                              hipStream_t stream) {
  (void)in_sizes; (void)n_in; (void)out_size; (void)ws_size;

  const float* x_h      = (const float*)d_in[0];
  const float* x_l      = (const float*)d_in[1];
  const float* bn_h_g   = (const float*)d_in[2];
  const float* bn_h_b   = (const float*)d_in[3];
  const float* bn_h_m   = (const float*)d_in[4];
  const float* bn_h_v   = (const float*)d_in[5];
  const float* kv_w     = (const float*)d_in[6];
  const float* kv_b     = (const float*)d_in[7];
  const float* bn_l_g   = (const float*)d_in[8];
  const float* bn_l_b   = (const float*)d_in[9];
  const float* bn_l_m   = (const float*)d_in[10];
  const float* bn_l_v   = (const float*)d_in[11];
  const float* reatt    = (const float*)d_in[12];

  // workspace layout (f32 region, then bf16 region)
  float* ws_f    = (float*)d_ws;
  float* scale_h = ws_f;             // 128
  float* shift_h = ws_f + 128;       // 128
  float* scale_l = ws_f + 256;       // 256
  float* shift_l = ws_f + 512;       // 256
  float* bias    = ws_f + 768;       // B_*K_ = 512
  float* p       = ws_f + 1280;      // 8*128*64 = 65536 floats
  float* ckf     = ws_f + 1280 + 65536;                    // 131072 floats
  __bf16* ckb = (__bf16*)(ckf + (size_t)B_ * K_ * CL_);    // 131072 bf16
  __bf16* cvb = ckb + (size_t)B_ * K_ * CL_;               // 131072 bf16

  ea_prep<<<1, 256, 0, stream>>>(bn_h_g, bn_h_b, bn_h_m, bn_h_v,
                                 bn_l_g, bn_l_b, bn_l_m, bn_l_v,
                                 scale_h, shift_h, scale_l, shift_l);

  ea_pool<<<(B_ * CH_ * K_) / 256, 256, 0, stream>>>(x_h, scale_h, shift_h, p);

  ea_kv<<<(B_ * 2 * CL_ * K_) / 256, 256, 0, stream>>>(p, kv_w, kv_b, scale_l,
                                                       ckb, ckf, cvb);

  ea_bias<<<(B_ * K_) / 256, 256, 0, stream>>>(ckf, shift_l, reatt, bias);

  dim3 grid(HWL_ / 128, B_);
  ea_main<<<grid, 256, 0, stream>>>(x_l, ckb, cvb, bias, (float*)d_out);
}
